// SimpleTTSModel_42339787604826
// MI455X (gfx1250) — matrix operand order repaired and checked by reading, not compiled, not run on hardware
//
#include <hip/hip_runtime.h>
#include <hip/hip_bf16.h>
#include <stdint.h>

// ---------------------------------------------------------------------------
// Types for CDNA5 WMMA (gfx1250, wave32)
// ---------------------------------------------------------------------------
typedef _Float16 h16;
typedef __attribute__((ext_vector_type(4)))  _Float16 v4h;
typedef __attribute__((ext_vector_type(8)))  _Float16 v8h;
typedef __attribute__((ext_vector_type(16))) _Float16 v16h;
typedef __attribute__((ext_vector_type(4)))  float    v4f;
typedef __attribute__((ext_vector_type(8)))  float    v8f;

#define LNUM   6
#define HDIM   512
#define NHEAD  8
#define FFDIM  2048
#define MELD   80
#define BATCH  16
#define TTXT   256
#define TDEC   512
#define DHEAD  64
#define GHEADS (BATCH*NHEAD)   /* 128 head-batches */
#define EPSLN  1e-5f

static inline int cdiv(int a, int b){ return (a + b - 1) / b; }

// vector-load-4 + convert to f16 (overloaded on source type)
__device__ __forceinline__ v4h ldcvt4(const float* p) {
    v4f t = *(const v4f*)p;
    v4h o; o[0] = (h16)t[0]; o[1] = (h16)t[1]; o[2] = (h16)t[2]; o[3] = (h16)t[3];
    return o;
}
__device__ __forceinline__ v4h ldcvt4(const h16* p) { return *(const v4h*)p; }

// ---------------------------------------------------------------------------
// WMMA GEMM:  C[M,N] = act( scale * A[M,K] x Wt[N,K]^T + bias[N] )
//  - A: fp32 or f16 (template), row-major, row stride lda
//  - Wt: f16, stored [N][K] (W transposed) so fragments are contiguous b128
//  - batched via blockIdx.z with element strides (0 strides => single GEMM)
//  - block = 256 threads = 8 waves; block tile 256x64; wave tile 32x64
//    -> 8 WMMAs per K-chunk per wave from 2 A-frags + 4 B-frags
//  - FULL=true: M%256==0, N%64==0, K%32==0 -> completely branch/check free
//  - software-pipelined: next K-chunk staged into registers during WMMA
// Fragment layouts per CDNA5 ISA 7.12.2:
//  A lane: row = lane&15, halves 0..7 -> K=(lane>>4)*8+0..7, 8..15 -> +16
//  B lane: col = lane&15, halves 0..7 -> K=(lane>>4)*8+0..7, 8..15 -> +16
//  C lane: element j -> m = j + (lane>>4)*8, n = lane&15
// ---------------------------------------------------------------------------
template<typename AT, bool FULL>
__global__ __launch_bounds__(256) void gemm_wmma(
    const AT* __restrict__ Abase, long long strideA,
    const h16* __restrict__ Bbase, long long strideB,
    const float* __restrict__ bias,
    float* __restrict__ Cbase, long long strideC,
    int M, int N, int K, int lda, float scale, int relu)
{
    const AT*  A  = Abase + (size_t)blockIdx.z * strideA;
    const h16* Wt = Bbase + (size_t)blockIdx.z * strideB;
    float*     C  = Cbase + (size_t)blockIdx.z * strideC;

    const int bm = blockIdx.y * 256;
    const int bn = blockIdx.x * 64;

    // +8 halfs pad keeps rows 16B aligned (40 halfs = 80B)
    __shared__ h16 sA[256][40];
    __shared__ h16 sB[64][40];

    const int tid  = threadIdx.x;
    const int lane = tid & 31;
    const int wave = tid >> 5;
    const int m0 = wave * 32;         // 8 waves along M; each wave owns all N
    const int frow = lane & 15;       // A row / B col within 16-tile
    const int ka   = (lane >> 4) * 8; // A K sub-offset
    const int kb   = (lane >> 4) * 8; // B K sub-offset

    // staging geometry: 8 threads per row (4 elems each), 32 rows per pass
    const int sc_ = (tid & 7) * 4;    // K offset within chunk
    const int sr_ = tid >> 3;         // base row 0..31

    v8f acc[2][4] = {};
    union Frag { v16h v; v8h h[2]; };

    v4h ra[8], rb[2];

    // stage one K-chunk into registers (vector fast path, scalar on edges)
    auto stage = [&](int kk) {
        #pragma unroll
        for (int it = 0; it < 8; ++it) {            // A: 256 rows
            int gm = bm + sr_ + it * 32;
            int gk = kk + sc_;
            if (FULL || (gm < M && gk + 3 < K)) {
                ra[it] = ldcvt4(&A[(size_t)gm * lda + gk]);
            } else {
                #pragma unroll
                for (int j = 0; j < 4; ++j)
                    ra[it][j] = (gm < M && gk + j < K) ? (h16)A[(size_t)gm * lda + gk + j]
                                                       : (h16)0.0f;
            }
        }
        #pragma unroll
        for (int it = 0; it < 2; ++it) {            // B: 64 rows
            int gn = bn + sr_ + it * 32;
            int gk = kk + sc_;
            if (FULL || (gn < N && gk + 3 < K)) {
                rb[it] = ldcvt4(&Wt[(size_t)gn * K + gk]);
            } else {
                #pragma unroll
                for (int j = 0; j < 4; ++j)
                    rb[it][j] = (gn < N && gk + j < K) ? Wt[(size_t)gn * K + gk + j]
                                                       : (h16)0.0f;
            }
        }
    };

    stage(0);
    for (int kk = 0; kk < K; kk += 32) {
        // commit staged registers to LDS
        #pragma unroll
        for (int it = 0; it < 8; ++it) *(v4h*)&sA[sr_ + it * 32][sc_] = ra[it];
        #pragma unroll
        for (int it = 0; it < 2; ++it) *(v4h*)&sB[sr_ + it * 32][sc_] = rb[it];
        __syncthreads();

        // prefetch next chunk while WMMAs run
        if (kk + 32 < K) stage(kk + 32);

        Frag fa[2], fb[4];
        #pragma unroll
        for (int mi = 0; mi < 2; ++mi) {
            fa[mi].h[0] = *(const v8h*)&sA[m0 + mi * 16 + frow][ka];
            fa[mi].h[1] = *(const v8h*)&sA[m0 + mi * 16 + frow][ka + 16];
        }
        #pragma unroll
        for (int ni = 0; ni < 4; ++ni) {
            fb[ni].h[0] = *(const v8h*)&sB[ni * 16 + frow][kb];
            fb[ni].h[1] = *(const v8h*)&sB[ni * 16 + frow][kb + 16];
        }
        #pragma unroll
        for (int mi = 0; mi < 2; ++mi)
            #pragma unroll
            for (int ni = 0; ni < 4; ++ni)
                acc[mi][ni] = __builtin_amdgcn_wmma_f32_16x16x32_f16(
                    false, fa[mi].v, false, fb[ni].v,
                    (short)0, acc[mi][ni], false, false);
        __syncthreads();
    }

    // epilogue: scale, bias, relu, store fp32
    #pragma unroll
    for (int mi = 0; mi < 2; ++mi) {
        #pragma unroll
        for (int ni = 0; ni < 4; ++ni) {
            int n = bn + ni * 16 + (lane & 15);
            if (!FULL && n >= N) continue;
            float bb = bias ? bias[n] : 0.0f;
            int mbase = bm + m0 + mi * 16 + (lane >> 4) * 8;
            #pragma unroll
            for (int j = 0; j < 8; ++j) {
                int m = mbase + j;
                if (FULL || m < M) {
                    float v = acc[mi][ni][j] * scale + bb;
                    if (relu) v = fmaxf(v, 0.0f);
                    C[(size_t)m * N + n] = v;
                }
            }
        }
    }
}

// ---------------------------------------------------------------------------
// Small helper kernels
// ---------------------------------------------------------------------------
__global__ void convert_wt(const float* __restrict__ W, h16* __restrict__ Wt,
                           int K, int N, int total)
{
    int i = blockIdx.x * 256 + threadIdx.x;
    if (i >= total) return;
    int n = i / K, k = i % K;
    Wt[(size_t)n * K + k] = (h16)W[(size_t)k * N + n];
}

__global__ void gather_embed(const int* __restrict__ tok, const float* __restrict__ emb,
                             float* __restrict__ x, int total)
{
    int i = blockIdx.x * 256 + threadIdx.x;
    if (i >= total) return;
    int r = i >> 9, c = i & 511;
    x[i] = emb[(size_t)tok[r] * HDIM + c];
}

__global__ void slice_mel(const float* __restrict__ mel, float* __restrict__ out, int total)
{
    int i = blockIdx.x * 256 + threadIdx.x;
    if (i >= total) return;
    int r = i / MELD, c = i % MELD;
    int b = r >> 9, t = r & 511;                 // keep 512 of 513 rows
    out[i] = mel[((size_t)(b * 513 + t)) * MELD + c];
}

// src [B*T, ld] fp32 -> dst f16 per head: [g][T][64] or transposed [g][64][T]
__global__ void pack_head(const float* __restrict__ src, int ld, int colOff,
                          h16* __restrict__ dst, int T, int trans, int total)
{
    int i = blockIdx.x * 256 + threadIdx.x;
    if (i >= total) return;
    int d = i & 63;
    int t = (i >> 6) % T;
    int g = i / (T * 64);
    int b = g >> 3, h = g & 7;
    float v = src[(size_t)(b * T + t) * ld + colOff + h * DHEAD + d];
    if (trans) dst[(size_t)g * DHEAD * T + (size_t)d * T + t] = (h16)v;
    else       dst[(size_t)g * T * DHEAD + (size_t)t * DHEAD + d] = (h16)v;
}

// O [g][T][64] fp32 -> out [B*T, HDIM]
__global__ void merge_heads(const float* __restrict__ O, float* __restrict__ out,
                            int T, int total)
{
    int i = blockIdx.x * 256 + threadIdx.x;
    if (i >= total) return;
    int d = i & 63;
    int t = (i >> 6) % T;
    int g = i / (T * 64);
    int b = g >> 3, h = g & 7;
    out[(size_t)(b * T + t) * HDIM + h * DHEAD + d] = O[i];
}

// row-wise softmax: S fp32 [rows, cols] -> P f16
__global__ __launch_bounds__(256) void softmax_rows(const float* __restrict__ S,
                                                    h16* __restrict__ P, int cols)
{
    int row = blockIdx.x, tid = threadIdx.x;
    const float* s = S + (size_t)row * cols;
    h16* p = P + (size_t)row * cols;
    __shared__ float red[256];
    float m = -3.0e38f;
    for (int c = tid; c < cols; c += 256) m = fmaxf(m, s[c]);
    red[tid] = m; __syncthreads();
    for (int st = 128; st; st >>= 1) { if (tid < st) red[tid] = fmaxf(red[tid], red[tid + st]); __syncthreads(); }
    m = red[0]; __syncthreads();
    float sum = 0.0f;
    for (int c = tid; c < cols; c += 256) sum += __expf(s[c] - m);
    red[tid] = sum; __syncthreads();
    for (int st = 128; st; st >>= 1) { if (tid < st) red[tid] += red[tid + st]; __syncthreads(); }
    float inv = 1.0f / red[0];
    for (int c = tid; c < cols; c += 256) p[c] = (h16)(__expf(s[c] - m) * inv);
}

// x[row] = LN(x[row] + r[row]) * g + b   (H = 512, 2 elems/thread, in place)
__global__ __launch_bounds__(256) void add_layernorm(float* __restrict__ x,
                                                     const float* __restrict__ r,
                                                     const float* __restrict__ gam,
                                                     const float* __restrict__ bet)
{
    int row = blockIdx.x, tid = threadIdx.x;
    size_t base = (size_t)row * HDIM;
    float v0 = x[base + tid]       + r[base + tid];
    float v1 = x[base + tid + 256] + r[base + tid + 256];
    __shared__ float red[256];
    red[tid] = v0 + v1; __syncthreads();
    for (int st = 128; st; st >>= 1) { if (tid < st) red[tid] += red[tid + st]; __syncthreads(); }
    float mean = red[0] * (1.0f / HDIM); __syncthreads();
    float d0 = v0 - mean, d1 = v1 - mean;
    red[tid] = d0 * d0 + d1 * d1; __syncthreads();
    for (int st = 128; st; st >>= 1) { if (tid < st) red[tid] += red[tid + st]; __syncthreads(); }
    float inv = rsqrtf(red[0] * (1.0f / HDIM) + EPSLN);
    x[base + tid]       = d0 * inv * gam[tid]       + bet[tid];
    x[base + tid + 256] = d1 * inv * gam[tid + 256] + bet[tid + 256];
}

// ---------------------------------------------------------------------------
// Host orchestration
// ---------------------------------------------------------------------------
extern "C" void kernel_launch(void* const* d_in, const int* in_sizes, int n_in,
                              void* d_out, int out_size, void* d_ws, size_t ws_size,
                              hipStream_t stream)
{
    (void)in_sizes; (void)n_in; (void)out_size; (void)ws_size;

    // ---- inputs (setup_inputs dict order, nested dicts flattened in order) ----
    const int*   tok      = (const int*)  d_in[0];
    const float* mel_t    = (const float*)d_in[1];
    const float* embed    = (const float*)d_in[2];
    const float* e_qkv_w  = (const float*)d_in[3];
    const float* e_qkv_b  = (const float*)d_in[4];
    const float* e_o_w    = (const float*)d_in[5];
    const float* e_o_b    = (const float*)d_in[6];
    const float* e_ln1_g  = (const float*)d_in[7];
    const float* e_ln1_b  = (const float*)d_in[8];
    const float* e_ln2_g  = (const float*)d_in[9];
    const float* e_ln2_b  = (const float*)d_in[10];
    const float* e_ff1_w  = (const float*)d_in[11];
    const float* e_ff1_b  = (const float*)d_in[12];
    const float* e_ff2_w  = (const float*)d_in[13];
    const float* e_ff2_b  = (const float*)d_in[14];
    const float* dd_qkv_w = (const float*)d_in[15];
    const float* dd_qkv_b = (const float*)d_in[16];
    const float* dd_o_w   = (const float*)d_in[17];
    const float* dd_o_b   = (const float*)d_in[18];
    const float* dd_ln1_g = (const float*)d_in[19];
    const float* dd_ln1_b = (const float*)d_in[20];
    const float* dd_ln2_g = (const float*)d_in[21];
    const float* dd_ln2_b = (const float*)d_in[22];
    const float* dd_ff1_w = (const float*)d_in[23];
    const float* dd_ff1_b = (const float*)d_in[24];
    const float* dd_ff2_w = (const float*)d_in[25];
    const float* dd_ff2_b = (const float*)d_in[26];
    const float* dd_cq_w  = (const float*)d_in[27];
    const float* dd_cq_b  = (const float*)d_in[28];
    const float* dd_co_w  = (const float*)d_in[29];
    const float* dd_co_b  = (const float*)d_in[30];
    const float* dd_ln3_g = (const float*)d_in[31];
    const float* dd_ln3_b = (const float*)d_in[32];
    const float* mel_w    = (const float*)d_in[33];
    const float* mel_b    = (const float*)d_in[34];
    const float* proj_w   = (const float*)d_in[35];
    const float* proj_b   = (const float*)d_in[36];
    const float* stop_w   = (const float*)d_in[37];
    const float* stop_b   = (const float*)d_in[38];
    float* out = (float*)d_out;

    const int Me = BATCH * TTXT;   // 4096 encoder tokens
    const int Md = BATCH * TDEC;   // 8192 decoder tokens

    // ---- workspace bump allocator ----
    char* wsp = (char*)d_ws;
    size_t off = 0;
    auto alloc = [&](size_t bytes) -> void* {
        size_t a = (off + 255) & ~(size_t)255;
        off = a + bytes;
        return wsp + a;
    };

    // f16 transposed weights [N][K]
    h16* wt_eqkv = (h16*)alloc((size_t)LNUM * 3 * HDIM * HDIM * 2);
    h16* wt_eo   = (h16*)alloc((size_t)LNUM * HDIM * HDIM * 2);
    h16* wt_ef1  = (h16*)alloc((size_t)LNUM * FFDIM * HDIM * 2);
    h16* wt_ef2  = (h16*)alloc((size_t)LNUM * HDIM * FFDIM * 2);
    h16* wt_dqkv = (h16*)alloc((size_t)LNUM * 3 * HDIM * HDIM * 2);
    h16* wt_do   = (h16*)alloc((size_t)LNUM * HDIM * HDIM * 2);
    h16* wt_df1  = (h16*)alloc((size_t)LNUM * FFDIM * HDIM * 2);
    h16* wt_df2  = (h16*)alloc((size_t)LNUM * HDIM * FFDIM * 2);
    h16* wt_dcq  = (h16*)alloc((size_t)LNUM * 3 * HDIM * HDIM * 2);
    h16* wt_dco  = (h16*)alloc((size_t)LNUM * HDIM * HDIM * 2);
    h16* wt_mel  = (h16*)alloc((size_t)HDIM * MELD * 2);
    h16* wt_proj = (h16*)alloc((size_t)MELD * HDIM * 2);
    h16* wt_stop = (h16*)alloc((size_t)HDIM * 2);

    // activations
    float* xb   = (float*)alloc((size_t)Me * HDIM * 4);         // encoder stream / memory
    float* yb   = (float*)alloc((size_t)Md * HDIM * 4);         // decoder stream
    float* t0   = (float*)alloc((size_t)Md * FFDIM * 4);        // qkv out / ff hidden
    float* t1   = (float*)alloc((size_t)Md * HDIM * 4);         // merged attn
    float* t2   = (float*)alloc((size_t)Md * HDIM * 4);         // proj / ffn out
    float* tkv  = (float*)alloc((size_t)Me * 2 * HDIM * 4);     // cross k,v
    float* melc = (float*)alloc((size_t)Md * MELD * 4);         // sliced mel
    h16*   Qh   = (h16*)  alloc((size_t)GHEADS * TDEC * DHEAD * 2);
    h16*   Kh   = (h16*)  alloc((size_t)GHEADS * TDEC * DHEAD * 2);
    h16*   Vth  = (h16*)  alloc((size_t)GHEADS * TDEC * DHEAD * 2);
    float* Oh   = (float*)alloc((size_t)GHEADS * TDEC * DHEAD * 4);
    float* sc   = (float*)alloc((size_t)GHEADS * TDEC * TDEC * 4);
    h16*   ph   = (h16*)  alloc((size_t)GHEADS * TDEC * TDEC * 2);

    // ---- launch helpers ----
    auto conv = [&](const float* W, h16* Wt, int K, int N) {
        int total = K * N;
        convert_wt<<<cdiv(total, 256), 256, 0, stream>>>(W, Wt, K, N, total);
    };
    auto gemm = [&](const float* A, const h16* Wt, const float* bias, float* C,
                    int M, int N, int K, int relu) {
        dim3 grid(cdiv(N, 64), cdiv(M, 256), 1);
        if ((M % 256 == 0) && (N % 64 == 0) && (K % 32 == 0))
            gemm_wmma<float, true ><<<grid, 256, 0, stream>>>(A, 0, Wt, 0, bias, C, 0,
                                                              M, N, K, K, 1.0f, relu);
        else
            gemm_wmma<float, false><<<grid, 256, 0, stream>>>(A, 0, Wt, 0, bias, C, 0,
                                                              M, N, K, K, 1.0f, relu);
    };
    auto bgemm = [&](const h16* A, const h16* Bt, float* C,
                     int M, int N, int K, float scale) {
        dim3 grid(cdiv(N, 64), cdiv(M, 256), GHEADS);
        gemm_wmma<h16, true><<<grid, 256, 0, stream>>>(A, (long long)M * K,
                                                       Bt, (long long)N * K,
                                                       nullptr, C, (long long)M * N,
                                                       M, N, K, K, scale, 0);
    };
    auto attn = [&](const float* qsrc, int qld, int qoff,
                    const float* kvsrc, int kvld, int koff, int voff,
                    int Tq, int Tk, float* merged) {
        int tq = GHEADS * Tq * DHEAD, tk = GHEADS * Tk * DHEAD;
        pack_head<<<cdiv(tq, 256), 256, 0, stream>>>(qsrc, qld, qoff, Qh, Tq, 0, tq);
        pack_head<<<cdiv(tk, 256), 256, 0, stream>>>(kvsrc, kvld, koff, Kh, Tk, 0, tk);
        pack_head<<<cdiv(tk, 256), 256, 0, stream>>>(kvsrc, kvld, voff, Vth, Tk, 1, tk);
        bgemm(Qh, Kh, sc, Tq, Tk, DHEAD, 0.125f);                 // 1/sqrt(64)
        softmax_rows<<<GHEADS * Tq, 256, 0, stream>>>(sc, ph, Tk);
        bgemm(ph, Vth, Oh, Tq, DHEAD, Tk, 1.0f);
        merge_heads<<<cdiv(tq, 256), 256, 0, stream>>>(Oh, merged, Tq, tq);
    };

    // ---- weight conversion (every call: deterministic) ----
    for (int l = 0; l < LNUM; ++l) {
        conv(e_qkv_w  + (size_t)l * HDIM * 3 * HDIM, wt_eqkv + (size_t)l * 3 * HDIM * HDIM, HDIM, 3 * HDIM);
        conv(e_o_w    + (size_t)l * HDIM * HDIM,     wt_eo   + (size_t)l * HDIM * HDIM,     HDIM, HDIM);
        conv(e_ff1_w  + (size_t)l * HDIM * FFDIM,    wt_ef1  + (size_t)l * FFDIM * HDIM,    HDIM, FFDIM);
        conv(e_ff2_w  + (size_t)l * FFDIM * HDIM,    wt_ef2  + (size_t)l * HDIM * FFDIM,    FFDIM, HDIM);
        conv(dd_qkv_w + (size_t)l * HDIM * 3 * HDIM, wt_dqkv + (size_t)l * 3 * HDIM * HDIM, HDIM, 3 * HDIM);
        conv(dd_o_w   + (size_t)l * HDIM * HDIM,     wt_do   + (size_t)l * HDIM * HDIM,     HDIM, HDIM);
        conv(dd_ff1_w + (size_t)l * HDIM * FFDIM,    wt_df1  + (size_t)l * FFDIM * HDIM,    HDIM, FFDIM);
        conv(dd_ff2_w + (size_t)l * FFDIM * HDIM,    wt_df2  + (size_t)l * HDIM * FFDIM,    FFDIM, HDIM);
        conv(dd_cq_w  + (size_t)l * HDIM * 3 * HDIM, wt_dcq  + (size_t)l * 3 * HDIM * HDIM, HDIM, 3 * HDIM);
        conv(dd_co_w  + (size_t)l * HDIM * HDIM,     wt_dco  + (size_t)l * HDIM * HDIM,     HDIM, HDIM);
    }
    conv(mel_w,  wt_mel,  MELD, HDIM);
    conv(proj_w, wt_proj, HDIM, MELD);
    conv(stop_w, wt_stop, HDIM, 1);

    // ---- encoder ----
    gather_embed<<<cdiv(Me * HDIM, 256), 256, 0, stream>>>(tok, embed, xb, Me * HDIM);
    for (int l = 0; l < LNUM; ++l) {
        gemm(xb, wt_eqkv + (size_t)l * 3 * HDIM * HDIM, e_qkv_b + (size_t)l * 3 * HDIM,
             t0, Me, 3 * HDIM, HDIM, 0);
        attn(t0, 3 * HDIM, 0, t0, 3 * HDIM, HDIM, 2 * HDIM, TTXT, TTXT, t1);
        gemm(t1, wt_eo + (size_t)l * HDIM * HDIM, e_o_b + (size_t)l * HDIM,
             t2, Me, HDIM, HDIM, 0);
        add_layernorm<<<Me, 256, 0, stream>>>(xb, t2, e_ln1_g + l * HDIM, e_ln1_b + l * HDIM);
        gemm(xb, wt_ef1 + (size_t)l * FFDIM * HDIM, e_ff1_b + (size_t)l * FFDIM,
             t0, Me, FFDIM, HDIM, 1);
        gemm(t0, wt_ef2 + (size_t)l * HDIM * FFDIM, e_ff2_b + (size_t)l * HDIM,
             t2, Me, HDIM, FFDIM, 0);
        add_layernorm<<<Me, 256, 0, stream>>>(xb, t2, e_ln2_g + l * HDIM, e_ln2_b + l * HDIM);
    }
    // xb is now encoder memory

    // ---- decoder ----
    slice_mel<<<cdiv(Md * MELD, 256), 256, 0, stream>>>(mel_t, melc, Md * MELD);
    gemm(melc, wt_mel, mel_b, yb, Md, HDIM, MELD, 0);
    for (int l = 0; l < LNUM; ++l) {
        // self attention
        gemm(yb, wt_dqkv + (size_t)l * 3 * HDIM * HDIM, dd_qkv_b + (size_t)l * 3 * HDIM,
             t0, Md, 3 * HDIM, HDIM, 0);
        attn(t0, 3 * HDIM, 0, t0, 3 * HDIM, HDIM, 2 * HDIM, TDEC, TDEC, t1);
        gemm(t1, wt_do + (size_t)l * HDIM * HDIM, dd_o_b + (size_t)l * HDIM,
             t2, Md, HDIM, HDIM, 0);
        add_layernorm<<<Md, 256, 0, stream>>>(yb, t2, dd_ln1_g + l * HDIM, dd_ln1_b + l * HDIM);
        // cross attention: q from y, k/v from encoder memory
        gemm(yb, wt_dcq + (size_t)l * 3 * HDIM * HDIM, dd_cq_b + (size_t)l * 3 * HDIM,
             t0, Md, HDIM, HDIM, 0);                                 // q (Wt rows 0..511)
        gemm(xb, wt_dcq + (size_t)l * 3 * HDIM * HDIM + (size_t)HDIM * HDIM,
             dd_cq_b + (size_t)l * 3 * HDIM + HDIM,
             tkv, Me, 2 * HDIM, HDIM, 0);                            // k,v (rows 512..1535)
        attn(t0, HDIM, 0, tkv, 2 * HDIM, 0, HDIM, TDEC, TTXT, t1);
        gemm(t1, wt_dco + (size_t)l * HDIM * HDIM, dd_co_b + (size_t)l * HDIM,
             t2, Md, HDIM, HDIM, 0);
        add_layernorm<<<Md, 256, 0, stream>>>(yb, t2, dd_ln2_g + l * HDIM, dd_ln2_b + l * HDIM);
        // FFN
        gemm(yb, wt_df1 + (size_t)l * FFDIM * HDIM, dd_ff1_b + (size_t)l * FFDIM,
             t0, Md, FFDIM, HDIM, 1);
        gemm(t0, wt_df2 + (size_t)l * HDIM * FFDIM, dd_ff2_b + (size_t)l * HDIM,
             t2, Md, HDIM, FFDIM, 0);
        add_layernorm<<<Md, 256, 0, stream>>>(yb, t2, dd_ln3_g + l * HDIM, dd_ln3_b + l * HDIM);
    }

    // ---- heads: mel_pred [B,512,80] then stop_pred [B,512,1] ----
    gemm(yb, wt_proj, proj_b, out, Md, MELD, HDIM, 0);
    gemm(yb, wt_stop, stop_b, out + (size_t)Md * MELD, Md, 1, HDIM, 0);
}
